// Sg2ScVAEModel_68985764708532
// MI455X (gfx1250) — matrix operand order repaired and checked by reading, not compiled, not run on hardware
//
#include <hip/hip_runtime.h>
#include <cstdint>
#include <cstddef>

// ---------------- model dimensions ----------------
#define EMB   128
#define HID   512
#define D2    256      // 2*EMB
#define NN    10000    // nodes
#define NE    40000    // edges

typedef unsigned short u16t;
typedef __attribute__((ext_vector_type(8)))  unsigned short us8;
typedef __attribute__((ext_vector_type(16))) unsigned short us16;
typedef __attribute__((ext_vector_type(16))) __bf16        v16bf;
typedef __attribute__((ext_vector_type(8)))  float         v8f;

__device__ __forceinline__ u16t f2bf(float f) {
  uint32_t u = __builtin_bit_cast(uint32_t, f);
  uint32_t r = (u + 0x7FFFu + ((u >> 16) & 1u)) >> 16;   // round-to-nearest-even
  return (u16t)r;
}
__device__ __forceinline__ float bf2f(u16t h) {
  uint32_t u = ((uint32_t)h) << 16;
  return __builtin_bit_cast(float, u);
}

// two 16B loads -> one 32B fragment (A: gap=16 halves, B: gap=16 halves)
__device__ __forceinline__ us16 ld2x8(const u16t* p, int gap) {
  us8 lo = *(const us8*)(p);
  us8 hi = *(const us8*)(p + gap);
  us16 r;
#pragma unroll
  for (int i = 0; i < 8; ++i) { r[i] = lo[i]; r[i + 8] = hi[i]; }
  return r;
}

__device__ __forceinline__ v8f wmma_bf16(const us16& a, const us16& b, v8f c) {
  return __builtin_amdgcn_wmma_f32_16x16x32_bf16(
      false, __builtin_bit_cast(v16bf, a), false, __builtin_bit_cast(v16bf, b),
      (short)0, c, false, false);
}

__device__ __forceinline__ void ldB4(const u16t* const brow[4], int k0, us16 b[4]) {
#pragma unroll
  for (int j = 0; j < 4; ++j) b[j] = ld2x8(brow[j] + k0, 16);
}
__device__ __forceinline__ void wmma4(const us16& a, const us16 b[4], v8f acc[4]) {
#pragma unroll
  for (int j = 0; j < 4; ++j) acc[j] = wmma_bf16(a, b[j], acc[j]);
}

// ---------------------------------------------------------------------
// Accumulate one contiguous A segment (klen elements, klen%64==0, >=64)
// against B columns kbase..kbase+klen.  K loop unrolled x2 into a
// ping-pong between two named fragment register sets: the clause of
// loads for step k+1 is in flight while the 4 wmmas of step k execute.
// Branch-free straight-line inner loop (scalar counter only).
// ---------------------------------------------------------------------
__device__ __forceinline__ void seg_accum(const u16t* __restrict__ aseg,
                                          const u16t* const brow[4], int kbase,
                                          int klen, v8f acc[4]) {
  us16 a0 = ld2x8(aseg, 16), a1;
  us16 b0[4], b1[4];
  ldB4(brow, kbase, b0);
  int k = 32;
  for (; k < klen - 32; k += 64) {
    a1 = ld2x8(aseg + k, 16);        ldB4(brow, kbase + k, b1);
    wmma4(a0, b0, acc);
    a0 = ld2x8(aseg + k + 32, 16);   ldB4(brow, kbase + k + 32, b0);
    wmma4(a1, b1, acc);
  }
  a1 = ld2x8(aseg + k, 16);          ldB4(brow, kbase + k, b1);
  wmma4(a0, b0, acc);
  wmma4(a1, b1, acc);
}

// =====================================================================
// WMMA GEMM:  C[M,N] = act(A[M,K] * W[K,N] + bias[N])
//   A: bf16 row-major lda;  W pre-transposed Wt[N,K] bf16;  C bf16 or f32.
// One wave computes a 16x64 tile (A fragment reused across 4 wmma).
// Requires M%16==0, N%64==0, K%64==0 (true for every GEMM here).
// =====================================================================
template <bool RELU, bool F32OUT>
__global__ __launch_bounds__(256) void k_gemm(
    const u16t* __restrict__ A, int lda,
    const u16t* __restrict__ Wt, const float* __restrict__ bias,
    void* __restrict__ Cv, int ldc, int coff, int M, int N, int K)
{
  const int wave = blockIdx.x * (blockDim.x >> 5) + (threadIdx.x >> 5);
  const int lane = threadIdx.x & 31;
  const int tiles_n = N >> 6;
  const int tm = wave / tiles_n;
  const int tn = wave - tm * tiles_n;
  if (tm * 16 >= M) return;
  const int half = lane >> 4;
  const int m16  = lane & 15;

  const u16t* arow = A + (size_t)(tm * 16 + m16) * lda + half * 8;
  const int nbase = tn * 64 + m16;
  const u16t* brow[4];
#pragma unroll
  for (int j = 0; j < 4; ++j) brow[j] = Wt + (size_t)(nbase + j * 16) * K + half * 8;

  v8f acc[4] = {};
  seg_accum(arow, brow, 0, K, acc);

#pragma unroll
  for (int j = 0; j < 4; ++j) {
    const int n = nbase + j * 16;
    const float bv = bias[n];
#pragma unroll
    for (int r = 0; r < 8; ++r) {
      const int mr = tm * 16 + half * 8 + r;
      float v = acc[j][r] + bv;
      if (RELU) v = fmaxf(v, 0.0f);
      if (F32OUT) ((float*)Cv)[(size_t)mr * ldc + coff + n] = v;
      else        ((u16t*)Cv)[(size_t)mr * ldc + coff + n] = f2bf(v);
    }
  }
}

// =====================================================================
// Fused gather + WMMA GEMM for the edge MLP (net1 first layer):
//   H[e, :] = relu( [obj[s[e]] | pred[e] | obj[o[e]]] @ W + b )
// The virtual A row is never materialized.  The three concat segments
// are visited in order as three branch-free pipelined sub-loops (din is
// a multiple of 64), so there is no per-step pointer selection at all.
// K = 3*din, M = NE, ldc = N, relu, bf16 out.
// =====================================================================
__global__ __launch_bounds__(256) void k_gemm_edge(
    const u16t* __restrict__ obj, int din,
    const u16t* __restrict__ pred, int ldp,
    const int* __restrict__ sidx, const int* __restrict__ oidx,
    const u16t* __restrict__ Wt, const float* __restrict__ bias,
    u16t* __restrict__ C, int N)
{
  const int K = 3 * din;
  const int wave = blockIdx.x * (blockDim.x >> 5) + (threadIdx.x >> 5);
  const int lane = threadIdx.x & 31;
  const int tiles_n = N >> 6;
  const int tm = wave / tiles_n;
  const int tn = wave - tm * tiles_n;
  if (tm * 16 >= NE) return;
  const int half = lane >> 4;
  const int m16  = lane & 15;
  const int row  = tm * 16 + m16;     // edge id for this lane
  const int nbase = tn * 64 + m16;

  const u16t* seg0 = obj  + (size_t)sidx[row] * din + half * 8;
  const u16t* seg1 = pred + (size_t)row * ldp       + half * 8;
  const u16t* seg2 = obj  + (size_t)oidx[row] * din + half * 8;
  const u16t* brow[4];
#pragma unroll
  for (int j = 0; j < 4; ++j) brow[j] = Wt + (size_t)(nbase + j * 16) * K + half * 8;

  v8f acc[4] = {};
  seg_accum(seg0, brow, 0,       din, acc);   // obj[s[e]]
  seg_accum(seg1, brow, din,     din, acc);   // pred[e]
  seg_accum(seg2, brow, 2 * din, din, acc);   // obj[o[e]]

#pragma unroll
  for (int j = 0; j < 4; ++j) {
    const int n = nbase + j * 16;
    const float bv = bias[n];
#pragma unroll
    for (int r = 0; r < 8; ++r) {
      const int mr = tm * 16 + half * 8 + r;
      C[(size_t)mr * N + n] = f2bf(fmaxf(acc[j][r] + bv, 0.0f));
    }
  }
}

// ---------------- auxiliary kernels ----------------

// W[K,N] f32 -> Wt[N,K] bf16 (coalesced writes)
__global__ void k_cvtT(const float* __restrict__ in, u16t* __restrict__ out, int K, int N) {
  long long idx = (long long)blockIdx.x * blockDim.x + threadIdx.x;
  if (idx >= (long long)K * N) return;
  int n = (int)(idx / K);
  int k = (int)(idx - (long long)n * K);
  out[idx] = f2bf(in[(size_t)k * N + n]);
}

__global__ void k_cvt(const float* __restrict__ in, u16t* __restrict__ out, long long n) {
  long long i = (long long)blockIdx.x * blockDim.x + threadIdx.x;
  if (i < n) out[i] = f2bf(in[i]);
}

__global__ void k_zero(float* __restrict__ p, long long n) {
  long long i = (long long)blockIdx.x * blockDim.x + threadIdx.x;
  if (i < n) p[i] = 0.0f;
}

__global__ void k_counts(const int* __restrict__ s, const int* __restrict__ o,
                         float* __restrict__ c, int ne) {
  int e = blockIdx.x * blockDim.x + threadIdx.x;
  if (e < ne) { atomicAdd(&c[s[e]], 1.0f); atomicAdd(&c[o[e]], 1.0f); }
}

__global__ void k_inv(const float* __restrict__ c, float* __restrict__ inv, int n) {
  int i = blockIdx.x * blockDim.x + threadIdx.x;
  if (i < n) inv[i] = 1.0f / fmaxf(c[i], 1.0f);
}

// ov_box init: [obj_emb_box[objs] | boxes_gt @ box_lin_w + b]  (K=6, scalar)
__global__ void k_init_box(const int* __restrict__ objs, const float* __restrict__ boxes,
                           const float* __restrict__ emb, const float* __restrict__ w,
                           const float* __restrict__ b, u16t* __restrict__ out, int nn) {
  long long idx = (long long)blockIdx.x * blockDim.x + threadIdx.x;
  if (idx >= (long long)nn * D2) return;
  int node = (int)(idx >> 8);
  int c = (int)(idx & 255);
  float v;
  if (c < EMB) v = emb[(size_t)objs[node] * EMB + c];
  else {
    int cc = c - EMB;
    v = b[cc];
#pragma unroll
    for (int j = 0; j < 6; ++j) v += boxes[(size_t)node * 6 + j] * w[j * EMB + cc];
  }
  out[idx] = f2bf(v);
}

// ov_shape[:, :128] = obj_emb_shape[objs]  (ld = 256)
__global__ void k_init_semb(const int* __restrict__ objs, const float* __restrict__ emb,
                            u16t* __restrict__ out, int nn) {
  long long idx = (long long)blockIdx.x * blockDim.x + threadIdx.x;
  if (idx >= (long long)nn * EMB) return;
  int node = (int)(idx >> 7);
  int c = (int)(idx & 127);
  out[(size_t)node * D2 + c] = f2bf(emb[(size_t)objs[node] * EMB + c]);
}

// pred embedding gather (f32 -> bf16)
__global__ void k_gpred(const int* __restrict__ p, const float* __restrict__ emb,
                        u16t* __restrict__ out, int d, int ne) {
  long long idx = (long long)blockIdx.x * blockDim.x + threadIdx.x;
  if (idx >= (long long)ne * d) return;
  int e = (int)(idx / d);
  int c = (int)(idx - (long long)e * d);
  out[idx] = f2bf(emb[(size_t)p[e] * d + c]);
}

// strided bf16 sub-matrix copy (16B chunks)
__global__ void k_copy(const u16t* __restrict__ src, int ldsrc, int soff,
                       u16t* __restrict__ dst, int lddst, int doff, int w, int rows) {
  const int chunks = w >> 3;
  long long idx = (long long)blockIdx.x * blockDim.x + threadIdx.x;
  if (idx >= (long long)rows * chunks) return;
  int r = (int)(idx / chunks);
  int c = (int)(idx - (long long)r * chunks) * 8;
  *(us8*)(dst + (size_t)r * lddst + doff + c) = *(const us8*)(src + (size_t)r * ldsrc + soff + c);
}

// pooled.at[s].add(t[:, :512]); pooled.at[o].add(t[:, 512+dout : 1024+dout])
__global__ void k_scatter(const u16t* __restrict__ t, int ldt, int dout,
                          const int* __restrict__ s, const int* __restrict__ o,
                          float* __restrict__ pooled, int ne) {
  long long idx = (long long)blockIdx.x * blockDim.x + threadIdx.x;
  if (idx >= (long long)ne * HID) return;
  int e = (int)(idx >> 9);
  int d = (int)(idx & 511);
  atomicAdd(&pooled[(size_t)s[e] * HID + d], bf2f(t[(size_t)e * ldt + d]));
  atomicAdd(&pooled[(size_t)o[e] * HID + d], bf2f(t[(size_t)e * ldt + HID + dout + d]));
}

// pooled / max(counts,1) -> bf16
__global__ void k_norm(const float* __restrict__ pooled, const float* __restrict__ inv,
                       u16t* __restrict__ out, long long n) {
  long long idx = (long long)blockIdx.x * blockDim.x + threadIdx.x;
  if (idx < n) out[idx] = f2bf(pooled[idx] * inv[idx >> 9]);
}

// =====================================================================
// host orchestration
// =====================================================================
static inline unsigned blocks1d(long long n, int t) { return (unsigned)((n + t - 1) / t); }

extern "C" void kernel_launch(void* const* d_in, const int* in_sizes, int n_in,
                              void* d_out, int out_size, void* d_ws, size_t ws_size,
                              hipStream_t stream)
{
  (void)in_sizes; (void)n_in; (void)out_size; (void)ws_size;

  const int*   objs        = (const int*)d_in[0];
  const int*   s_idx       = (const int*)d_in[1];
  const int*   p_idx       = (const int*)d_in[2];
  const int*   o_idx       = (const int*)d_in[3];
  const float* boxes_gt    = (const float*)d_in[4];
  const float* shapes_gt   = (const float*)d_in[5];
  // params, recursive insertion order of the setup_inputs() dict
  const float* obj_emb_box   = (const float*)d_in[6];
  const float* obj_emb_shape = (const float*)d_in[7];
  const float* pred_emb_box  = (const float*)d_in[8];
  const float* pred_emb_shape= (const float*)d_in[9];
  const float* box_lin_w   = (const float*)d_in[10];
  const float* box_lin_b   = (const float*)d_in[11];
  const float* shape_lin_b = (const float*)d_in[13];

  char* wsp = (char*)d_ws;
  auto alloc = [&](size_t bytes) -> void* {
    void* p = (void*)wsp; wsp += (bytes + 255) & ~(size_t)255; return p;
  };
  auto cvtT = [&](const void* w, int K, int N) -> const u16t* {
    u16t* o = (u16t*)alloc((size_t)K * N * sizeof(u16t));
    k_cvtT<<<blocks1d((long long)K * N, 256), 256, 0, stream>>>((const float*)w, o, K, N);
    return o;
  };
  auto gemm = [&](const u16t* A, int lda, const u16t* Wt, const float* bias,
                  void* C, int ldc, int coff, int M, int N, int K, bool relu, bool f32out) {
    long long waves = (long long)(M / 16) * (N / 64);
    unsigned blk = blocks1d(waves * 32, 256);
    if      ( relu && !f32out) k_gemm<true ,false><<<blk,256,0,stream>>>(A,lda,Wt,bias,C,ldc,coff,M,N,K);
    else if (!relu && !f32out) k_gemm<false,false><<<blk,256,0,stream>>>(A,lda,Wt,bias,C,ldc,coff,M,N,K);
    else if ( relu &&  f32out) k_gemm<true ,true ><<<blk,256,0,stream>>>(A,lda,Wt,bias,C,ldc,coff,M,N,K);
    else                       k_gemm<false,true ><<<blk,256,0,stream>>>(A,lda,Wt,bias,C,ldc,coff,M,N,K);
  };
  auto gemm_edge = [&](const u16t* obj, int din, const u16t* pred, int ldp,
                       const u16t* Wt, const float* bias, u16t* C, int N) {
    long long waves = (long long)(NE / 16) * (N / 64);
    k_gemm_edge<<<blocks1d(waves * 32, 256), 256, 0, stream>>>(
        obj, din, pred, ldp, s_idx, o_idx, Wt, bias, C, N);
  };

  struct GW { const u16t *w1a,*w1b,*w2a,*w2b; const float *b1a,*b1b,*b2a,*b2b; };
  GW gb[5], gs[5], gh[3];
  for (int L = 0; L < 5; ++L) {            // gconv_box: din=dout=256
    int base = 14 + L * 8;
    gb[L].w1a = cvtT(d_in[base+0], 3*D2, HID);     gb[L].b1a = (const float*)d_in[base+1];
    gb[L].w1b = cvtT(d_in[base+2], HID, 2*HID+D2); gb[L].b1b = (const float*)d_in[base+3];
    gb[L].w2a = cvtT(d_in[base+4], HID, HID);      gb[L].b2a = (const float*)d_in[base+5];
    gb[L].w2b = cvtT(d_in[base+6], HID, D2);       gb[L].b2b = (const float*)d_in[base+7];
  }
  for (int L = 0; L < 5; ++L) {            // gconv_shape
    int base = 54 + L * 8;
    gs[L].w1a = cvtT(d_in[base+0], 3*D2, HID);     gs[L].b1a = (const float*)d_in[base+1];
    gs[L].w1b = cvtT(d_in[base+2], HID, 2*HID+D2); gs[L].b1b = (const float*)d_in[base+3];
    gs[L].w2a = cvtT(d_in[base+4], HID, HID);      gs[L].b2a = (const float*)d_in[base+5];
    gs[L].w2b = cvtT(d_in[base+6], HID, D2);       gs[L].b2b = (const float*)d_in[base+7];
  }
  for (int L = 0; L < 3; ++L) {            // gconv_shared: din=dout=512
    int base = 94 + L * 8;
    gh[L].w1a = cvtT(d_in[base+0], 3*HID, HID);    gh[L].b1a = (const float*)d_in[base+1];
    gh[L].w1b = cvtT(d_in[base+2], HID, 3*HID);    gh[L].b1b = (const float*)d_in[base+3];
    gh[L].w2a = cvtT(d_in[base+4], HID, HID);      gh[L].b2a = (const float*)d_in[base+5];
    gh[L].w2b = cvtT(d_in[base+6], HID, HID);      gh[L].b2b = (const float*)d_in[base+7];
  }
  const u16t* bmv_w0 = cvtT(d_in[118], D2, HID); const float* bmv_b0 = (const float*)d_in[119];
  const u16t* bmv_w1 = cvtT(d_in[120], HID, D2); const float* bmv_b1 = (const float*)d_in[121];
  const u16t* bm_w   = cvtT(d_in[122], D2, EMB); const float* bm_b   = (const float*)d_in[123];
  const u16t* bv_w   = cvtT(d_in[124], D2, EMB); const float* bv_b   = (const float*)d_in[125];
  const u16t* smv_w0 = cvtT(d_in[126], D2, HID); const float* smv_b0 = (const float*)d_in[127];
  const u16t* smv_w1 = cvtT(d_in[128], HID, D2); const float* smv_b1 = (const float*)d_in[129];
  const u16t* sm_w   = cvtT(d_in[130], D2, EMB); const float* sm_b   = (const float*)d_in[131];
  const u16t* sv_w   = cvtT(d_in[132], D2, EMB); const float* sv_b   = (const float*)d_in[133];
  const u16t* slin_wt= cvtT(d_in[12],  EMB, EMB);   // shape_lin.w (128x128)

  // activations (bf16 except pooled accumulators)
  u16t*  HBUF   = (u16t*)alloc((size_t)NE * HID * 2);
  u16t*  TBUF   = (u16t*)alloc((size_t)NE * 1536 * 2);
  u16t*  PRED   = (u16t*)alloc((size_t)NE * HID * 2);
  u16t*  PV     = (u16t*)alloc((size_t)NE * HID * 2);
  float* POOLED = (float*)alloc((size_t)NN * HID * 4);
  u16t*  PN     = (u16t*)alloc((size_t)NN * HID * 2);
  u16t*  GBUF   = (u16t*)alloc((size_t)NN * HID * 2);
  u16t*  OBJA   = (u16t*)alloc((size_t)NN * HID * 2);
  u16t*  OBJB   = (u16t*)alloc((size_t)NN * HID * 2);
  u16t*  OV     = (u16t*)alloc((size_t)NN * HID * 2);
  float* COUNTS = (float*)alloc((size_t)NN * 4);
  float* INVC   = (float*)alloc((size_t)NN * 4);
  u16t*  SHB    = (u16t*)alloc((size_t)NN * EMB * 2);

  // edge counts: identical for every layer -> compute once
  k_zero  <<<blocks1d(NN, 256), 256, 0, stream>>>(COUNTS, NN);
  k_counts<<<blocks1d(NE, 256), 256, 0, stream>>>(s_idx, o_idx, COUNTS, NE);
  k_inv   <<<blocks1d(NN, 256), 256, 0, stream>>>(COUNTS, INVC, NN);
  k_cvt   <<<blocks1d((long long)NN*EMB, 256), 256, 0, stream>>>(shapes_gt, SHB, (long long)NN*EMB);

  auto run_gconv = [&](const u16t* obj_in, int din, const u16t* pred, int ldp,
                       const GW& w, int dout, u16t* obj_out, int out_ld, int out_coff,
                       bool extract_pred) {
    const int n2 = 2 * HID + dout;
    gemm_edge(obj_in, din, pred, ldp, w.w1a, w.b1a, HBUF, HID);      // fused gather+GEMM
    gemm(HBUF, HID, w.w1b, w.b1b, TBUF, n2, 0, NE, n2, HID, true, false);
    if (extract_pred)
      k_copy<<<blocks1d((long long)NE * (dout >> 3), 256), 256, 0, stream>>>(
          TBUF, n2, HID, PRED, dout, 0, dout, NE);
    k_zero   <<<blocks1d((long long)NN*HID, 256), 256, 0, stream>>>(POOLED, (long long)NN*HID);
    k_scatter<<<blocks1d((long long)NE*HID, 256), 256, 0, stream>>>(TBUF, n2, dout, s_idx, o_idx, POOLED, NE);
    k_norm   <<<blocks1d((long long)NN*HID, 256), 256, 0, stream>>>(POOLED, INVC, PN, (long long)NN*HID);
    gemm(PN,   HID, w.w2a, w.b2a, GBUF,    HID,    0,        NN, HID,  HID, true, false);
    gemm(GBUF, HID, w.w2b, w.b2b, obj_out, out_ld, out_coff, NN, dout, HID, true, false);
  };

  // ---- box chain ----
  k_init_box<<<blocks1d((long long)NN*D2, 256), 256, 0, stream>>>(
      objs, boxes_gt, obj_emb_box, box_lin_w, box_lin_b, OBJA, NN);
  k_gpred<<<blocks1d((long long)NE*D2, 256), 256, 0, stream>>>(p_idx, pred_emb_box, PRED, D2, NE);
  run_gconv(OBJA, D2, PRED, D2, gb[0], D2, OBJB, D2,  0, true);
  run_gconv(OBJB, D2, PRED, D2, gb[1], D2, OBJA, D2,  0, true);
  run_gconv(OBJA, D2, PRED, D2, gb[2], D2, OBJB, D2,  0, true);
  run_gconv(OBJB, D2, PRED, D2, gb[3], D2, OBJA, D2,  0, true);
  run_gconv(OBJA, D2, PRED, D2, gb[4], D2, OV,   HID, 0, true);
  k_copy<<<blocks1d((long long)NE*(D2>>3), 256), 256, 0, stream>>>(PRED, D2, 0, PV, HID, 0, D2, NE);

  // ---- shape chain ----
  k_init_semb<<<blocks1d((long long)NN*EMB, 256), 256, 0, stream>>>(objs, obj_emb_shape, OBJA, NN);
  gemm(SHB, EMB, slin_wt, shape_lin_b, OBJA, D2, EMB, NN, EMB, EMB, false, false);
  k_gpred<<<blocks1d((long long)NE*D2, 256), 256, 0, stream>>>(p_idx, pred_emb_shape, PRED, D2, NE);
  run_gconv(OBJA, D2, PRED, D2, gs[0], D2, OBJB, D2,  0,  true);
  run_gconv(OBJB, D2, PRED, D2, gs[1], D2, OBJA, D2,  0,  true);
  run_gconv(OBJA, D2, PRED, D2, gs[2], D2, OBJB, D2,  0,  true);
  run_gconv(OBJB, D2, PRED, D2, gs[3], D2, OBJA, D2,  0,  true);
  run_gconv(OBJA, D2, PRED, D2, gs[4], D2, OV,   HID, D2, true);
  k_copy<<<blocks1d((long long)NE*(D2>>3), 256), 256, 0, stream>>>(PRED, D2, 0, PV, HID, D2, D2, NE);

  // ---- shared chain (din = dout = 512) ----
  run_gconv(OV,   HID, PV,   HID, gh[0], HID, OBJA, HID, 0, true);
  run_gconv(OBJA, HID, PRED, HID, gh[1], HID, OBJB, HID, 0, true);
  run_gconv(OBJB, HID, PRED, HID, gh[2], HID, OBJA, HID, 0, false);

  // ---- output heads (f32 results straight into d_out) ----
  float* out0 = (float*)d_out;
  gemm(OBJA,      HID, bmv_w0, bmv_b0, PN,   HID, 0, NN, HID, D2,  true,  false);
  gemm(PN,        HID, bmv_w1, bmv_b1, GBUF, D2,  0, NN, D2,  HID, true,  false);
  gemm(GBUF,      D2,  bm_w,   bm_b,   out0,                      EMB, 0, NN, EMB, D2, false, true);
  gemm(GBUF,      D2,  bv_w,   bv_b,   out0 + (size_t)NN*EMB,     EMB, 0, NN, EMB, D2, false, true);
  gemm(OBJA + D2, HID, smv_w0, smv_b0, PN,   HID, 0, NN, HID, D2,  true,  false);
  gemm(PN,        HID, smv_w1, smv_b1, GBUF, D2,  0, NN, D2,  HID, true,  false);
  gemm(GBUF,      D2,  sm_w,   sm_b,   out0 + 2*(size_t)NN*EMB,   EMB, 0, NN, EMB, D2, false, true);
  gemm(GBUF,      D2,  sv_w,   sv_b,   out0 + 3*(size_t)NN*EMB,   EMB, 0, NN, EMB, D2, false, true);
}
